// GATFrontEnd_68470368633488
// MI455X (gfx1250) — matrix operand order repaired and checked by reading, not compiled, not run on hardware
//
#include <hip/hip_runtime.h>
#include <hip/hip_bf16.h>

typedef __attribute__((ext_vector_type(16))) __bf16 v16bf;
typedef __attribute__((ext_vector_type(8)))  __bf16 v8bf;
typedef __attribute__((ext_vector_type(8)))  float  v8f;
typedef int i32x4 __attribute__((vector_size(16)));   // payload type of async b128 builtin

#define LN_EPS 1e-5f
#define NEG_SLOPE 0.2f
#define ENC_NEG_INF 0x007FFFFFu

#if __has_builtin(__builtin_amdgcn_global_load_async_to_lds_b128) && \
    __has_builtin(__builtin_amdgcn_s_wait_asynccnt)
#define HAVE_ASYNC_LDS 1
#else
#define HAVE_ASYNC_LDS 0
#endif

// ---------- helpers ----------
__device__ __forceinline__ __bf16 f2bf(float f) {
  unsigned u = __float_as_uint(f);
  unsigned r = u + 0x7FFFu + ((u >> 16) & 1u);   // round-to-nearest-even
  unsigned short hs = (unsigned short)(r >> 16);
  return __builtin_bit_cast(__bf16, hs);
}
// order-preserving float<->uint for atomicMax-based segment max
__device__ __forceinline__ unsigned fenc(float f) {
  unsigned u = __float_as_uint(f);
  return u ^ ((u >> 31) ? 0xFFFFFFFFu : 0x80000000u);
}
__device__ __forceinline__ float fdec(unsigned u) {
  unsigned b = (u & 0x80000000u) ? (u ^ 0x80000000u) : ~u;
  return __uint_as_float(b);
}
__device__ __forceinline__ v16bf join16(v8bf lo, v8bf hi) {
  return __builtin_shufflevector(lo, hi, 0, 1, 2, 3, 4, 5, 6, 7,
                                 8, 9, 10, 11, 12, 13, 14, 15);
}

// ---------- cooperative LDS staging of packed bf16 weights ----------
// issue: every thread copies 8-bf16 (16B) chunks; commit: wait + barrier.
__device__ __forceinline__ void stage_issue(const __bf16* g, __bf16* l, int total) {
#if HAVE_ASYNC_LDS
  for (int o = (int)threadIdx.x * 8; o < total; o += (int)blockDim.x * 8) {
    __builtin_amdgcn_global_load_async_to_lds_b128(
        (__attribute__((address_space(1))) i32x4*)(g + o),
        (__attribute__((address_space(3))) i32x4*)(l + o), 0, 0);
  }
#else
  for (int o = (int)threadIdx.x * 8; o < total; o += (int)blockDim.x * 8)
    *(v8bf*)(l + o) = *(const v8bf*)(g + o);
#endif
}
__device__ __forceinline__ void stage_commit() {
#if HAVE_ASYNC_LDS
  __builtin_amdgcn_s_wait_asynccnt(0);
#endif
  __syncthreads();
}

// ---------- generic fill ----------
__global__ void k_fill_u32(unsigned* p, unsigned v, long long n) {
  long long i = (long long)blockIdx.x * blockDim.x + threadIdx.x;
  long long st = (long long)gridDim.x * blockDim.x;
  for (; i < n; i += st) p[i] = v;
}

// ---------- h0 (bf16) = node_emb[node_ids] + type_emb[node_type] ----------
__global__ void k_h0(const int* __restrict__ ids, const int* __restrict__ types,
                     const float* __restrict__ nodeW, const float* __restrict__ typeW,
                     __bf16* __restrict__ h0, int N) {
  int i = blockIdx.x * blockDim.x + threadIdx.x;
  if (i >= N * 64) return;
  int n = i >> 6, d = i & 63;
  h0[i] = f2bf(nodeW[(size_t)ids[n] * 64 + d] + typeW[(size_t)types[n] * 64 + d]);
}

// ---------- mean of edge_attr (single block reduction) ----------
__global__ void k_eamean(const float* __restrict__ ea, float* __restrict__ out, int E) {
  __shared__ float red[256];
  float s = 0.f;
  for (int i = threadIdx.x; i < E; i += blockDim.x) s += ea[i];
  red[threadIdx.x] = s;
  __syncthreads();
  for (int off = 128; off; off >>= 1) {
    if ((int)threadIdx.x < off) red[threadIdx.x] += red[threadIdx.x + off];
    __syncthreads();
  }
  if (threadIdx.x == 0) out[0] = red[0] / (float)E;
}

// ---------- pack f32 weight [K,Nc] into WMMA B-fragment lane order (bf16) ----------
// packed[(ks*nt + tn)*512 + lane*16 + e] = B[ks*32 + (lane>=16?8:0) + e + (e>=8?8:0)][tn*16 + lane%16]
__global__ void k_pack_w(const float* __restrict__ B, __bf16* __restrict__ Bp,
                         int K, int Nc) {
  int idx = blockIdx.x * blockDim.x + threadIdx.x;
  if (idx >= K * Nc) return;
  int e    = idx & 15;
  int l    = (idx >> 4) & 31;
  int tile = idx >> 9;
  int nt = Nc >> 4;
  int tn = tile % nt, ks = tile / nt;
  int k = ks * 32 + ((l >= 16) ? 8 : 0) + e + ((e >= 8) ? 8 : 0);
  int n = tn * 16 + (l & 15);
  Bp[idx] = f2bf(B[(size_t)k * Nc + n]);
}

// ---------- WMMA GEMM: C[M,Nc] = A[M,K] @ B[K,Nc] (+bias) ----------
// A: row-major bf16. Bp: pre-packed bf16 fragments, staged to LDS (async). C: f32.
// One wave computes a 16x64 output block (4 N-tiles), reusing the A fragment 4x.
__global__ void k_gemm_wmma(const __bf16* __restrict__ A, const __bf16* __restrict__ Bp,
                            const float* __restrict__ bias, float* __restrict__ C,
                            int M, int K, int Nc) {
  __shared__ __bf16 sB[16384];                 // K*Nc <= 16384 for all our GEMMs (32 KB)
  stage_issue(Bp, sB, K * Nc);
  stage_commit();                              // all threads reach barrier before any exit

  int wid  = (blockIdx.x * blockDim.x + threadIdx.x) >> 5;
  int lane = threadIdx.x & 31;
  int mt = M >> 4, nt4 = Nc >> 6;
  if (wid >= mt * nt4) return;                 // wave-uniform exit (EXEC all-1 for WMMA)
  int tm = wid / nt4, tg = wid % nt4;
  int lm = lane & 15;
  int hi = lane >> 4;
  int nt = Nc >> 4;
  // A-fragment: lane row = tm*16+lm; K runs [akb, akb+8) and [akb+16, akb+24)
  const __bf16* Ar = A + (size_t)(tm * 16 + lm) * K + (hi ? 8 : 0);
  v8f acc[4] = {{}, {}, {}, {}};
  int ksteps = K >> 5;
  for (int ks = 0; ks < ksteps; ++ks) {
    v8bf a0 = *(const v8bf*)(Ar + ks * 32);
    v8bf a1 = *(const v8bf*)(Ar + ks * 32 + 16);
    v16bf a = join16(a0, a1);
#pragma unroll
    for (int t = 0; t < 4; ++t) {
      int tn = tg * 4 + t;
      v16bf b = *(const v16bf*)(sB + ((ks * nt + tn) * 32 + lane) * 16);
      acc[t] = __builtin_amdgcn_wmma_f32_16x16x32_bf16(false, a, false, b, (short)0,
                                                       acc[t], false, false);
    }
  }
  int rbase = tm * 16 + (hi ? 8 : 0);          // D rows: v + 8*half
#pragma unroll
  for (int t = 0; t < 4; ++t) {
    int bcol = (tg * 4 + t) * 16 + lm;
    float bb = bias ? bias[bcol] : 0.f;
#pragma unroll
    for (int v = 0; v < 8; ++v)
      C[(size_t)(rbase + v) * Nc + bcol] = acc[t][v] + bb;
  }
}

// ---------- GATv2 attention logits + segment max (wave per edge) ----------
__global__ void k_att_logits(const float* __restrict__ xl, const float* __restrict__ xr,
                             const float* __restrict__ We, const float* __restrict__ att,
                             const int* __restrict__ ei, const float* __restrict__ ea,
                             const float* __restrict__ eamean,
                             unsigned* __restrict__ maxb, float* __restrict__ sc,
                             int E, int N, int H) {
  int wid  = (blockIdx.x * blockDim.x + threadIdx.x) >> 5;
  int lane = threadIdx.x & 31;
  int Etot = E + N;
  if (wid >= Etot) return;
  int s, d; float e_a;
  if (wid < E) { s = ei[wid]; d = ei[E + wid]; e_a = ea[wid]; }
  else         { s = d = wid - E;              e_a = eamean[0]; }   // self loop, mean fill
  int HC = H * 64;
  const float* xs = xl + (size_t)s * HC;
  const float* xd = xr + (size_t)d * HC;
  for (int h = 0; h < H; ++h) {
    float p = 0.f;
#pragma unroll
    for (int i = 0; i < 2; ++i) {
      int c = h * 64 + lane + (i << 5);
      float v = xs[c] + xd[c] + e_a * We[c];
      v = v > 0.f ? v : v * NEG_SLOPE;        // leaky_relu
      p += v * att[c];
    }
    for (int off = 16; off; off >>= 1) p += __shfl_xor(p, off, 32);
    if (lane == 0) {
      sc[(size_t)wid * H + h] = p;
      atomicMax(&maxb[(size_t)d * H + h], fenc(p));
    }
  }
}

// ---------- exp(sc - max[dst]) and segment sum ----------
__global__ void k_att_exp(const int* __restrict__ ei, const unsigned* __restrict__ maxb,
                          float* __restrict__ sc, float* __restrict__ den,
                          int E, int N, int hshift) {
  int H = 1 << hshift;
  long long i   = (long long)blockIdx.x * blockDim.x + threadIdx.x;
  long long tot = (long long)(E + N) << hshift;
  if (i >= tot) return;
  int e = (int)(i >> hshift), h = (int)(i & (H - 1));
  int d = (e < E) ? ei[E + e] : (e - E);
  float m = fdec(maxb[(size_t)d * H + h]);
  float v = __expf(sc[i] - m);
  sc[i] = v;                                   // overwrite logits with exp
  atomicAdd(&den[(size_t)d * H + h], v);
}

// ---------- alpha-weighted scatter: macc[dst] += alpha * xl[src] ----------
__global__ void k_att_scatter(const float* __restrict__ xl, const int* __restrict__ ei,
                              const float* __restrict__ ex, const float* __restrict__ den,
                              float* __restrict__ macc, int E, int N, int H) {
  int wid  = (blockIdx.x * blockDim.x + threadIdx.x) >> 5;
  int lane = threadIdx.x & 31;
  int Etot = E + N;
  if (wid >= Etot) return;
  int s, d;
  if (wid < E) { s = ei[wid]; d = ei[E + wid]; } else { s = d = wid - E; }
  int HC = H * 64;
  const float* xs = xl + (size_t)s * HC;
  float* md = macc + (size_t)d * HC;
  for (int idx = lane; idx < HC; idx += 32) {
    int h = idx >> 6;
    float alpha = ex[(size_t)wid * H + h] / den[(size_t)d * H + h];
    atomicAdd(&md[idx], alpha * xs[idx]);
  }
}

// ---------- h = LayerNorm(res + elu(macc + bias)) -> bf16 (wave per node) ----------
__global__ void k_elu_res_ln(const float* __restrict__ res, const float* __restrict__ macc,
                             const float* __restrict__ bias, const float* __restrict__ g,
                             const float* __restrict__ b, __bf16* __restrict__ out,
                             int N, int D) {
  int wid  = (blockIdx.x * blockDim.x + threadIdx.x) >> 5;
  int lane = threadIdx.x & 31;
  if (wid >= N) return;
  const float* r  = res  + (size_t)wid * D;
  const float* mm = macc + (size_t)wid * D;
  float vals[8];
  int cnt = D >> 5;
  float sum = 0.f;
  for (int i = 0; i < cnt; ++i) {
    int idx = lane + (i << 5);
    float x = mm[idx] + bias[idx];
    x = x > 0.f ? x : (__expf(x) - 1.f);       // elu
    x += r[idx];
    vals[i] = x; sum += x;
  }
  for (int off = 16; off; off >>= 1) sum += __shfl_xor(sum, off, 32);
  float mu = sum / (float)D;
  float var = 0.f;
  for (int i = 0; i < cnt; ++i) { float t = vals[i] - mu; var += t * t; }
  for (int off = 16; off; off >>= 1) var += __shfl_xor(var, off, 32);
  float rstd = rsqrtf(var / (float)D + LN_EPS);
  __bf16* o = out + (size_t)wid * D;
  for (int i = 0; i < cnt; ++i) {
    int idx = lane + (i << 5);
    o[idx] = f2bf((vals[i] - mu) * rstd * g[idx] + b[idx]);
  }
}

// ---------- fused edge MLP head via WMMA: 16 edges per wave ----------
// z = [h2[s], h2[t]] (K=128, bf16); hidden = relu(z@W1+b1) (64); out = hidden@W2+b2.
// Both heads (exist, amount) share the A fragment; W1's pre-packed + LDS-staged.
__global__ void k_edge_mlp(const __bf16* __restrict__ h2, const int* __restrict__ ei,
                           const __bf16* __restrict__ Wex1p, const float* __restrict__ bex1,
                           const float* __restrict__ Wex2, const float* __restrict__ bex2,
                           const __bf16* __restrict__ Wam1p, const float* __restrict__ bam1,
                           const float* __restrict__ Wam2, const float* __restrict__ bam2,
                           float* __restrict__ out, int E) {
  __shared__ __bf16 sW[16384];                 // [0,8192): Wex1p  [8192,16384): Wam1p
  stage_issue(Wex1p, sW, 8192);
  stage_issue(Wam1p, sW + 8192, 8192);
  stage_commit();

  int wid  = (blockIdx.x * blockDim.x + threadIdx.x) >> 5;
  int lane = threadIdx.x & 31;
  int e0 = wid * 16;
  if (e0 >= E) return;
  int lm = lane & 15;
  int hi = lane >> 4;
  int ed = e0 + lm;                              // edge row this lane loads for A
  const __bf16* zs = h2 + (size_t)ei[ed] * 64;   // src half of z
  const __bf16* zd = h2 + (size_t)ei[E + ed] * 64;  // dst half of z
  int akb = hi ? 8 : 0;
  v8f accE[4] = {{}, {}, {}, {}};
  v8f accA[4] = {{}, {}, {}, {}};
#pragma unroll
  for (int ks = 0; ks < 4; ++ks) {               // K = 128
    int b0 = ks * 32 + akb;                      // both runs are 8-aligned, so each
    int b1 = b0 + 16;                            // resolves wholly to zs or zd
    v8bf r0 = *(const v8bf*)((b0 < 64) ? (zs + b0) : (zd + b0 - 64));
    v8bf r1 = *(const v8bf*)((b1 < 64) ? (zs + b1) : (zd + b1 - 64));
    v16bf a = join16(r0, r1);
#pragma unroll
    for (int t = 0; t < 4; ++t) {                // hidden cols 16t..16t+15
      int po = ((ks * 4 + t) * 32 + lane) * 16;
      v16bf bE = *(const v16bf*)(sW + po);
      v16bf bA = *(const v16bf*)(sW + 8192 + po);
      accE[t] = __builtin_amdgcn_wmma_f32_16x16x32_bf16(false, a, false, bE, (short)0,
                                                        accE[t], false, false);
      accA[t] = __builtin_amdgcn_wmma_f32_16x16x32_bf16(false, a, false, bA, (short)0,
                                                        accA[t], false, false);
    }
  }
  // second stage: relu(hidden + b1) . W2, reduced across the 16 lanes per half
  float rowE[8] = {0, 0, 0, 0, 0, 0, 0, 0};
  float rowA[8] = {0, 0, 0, 0, 0, 0, 0, 0};
#pragma unroll
  for (int t = 0; t < 4; ++t) {
    int c = t * 16 + lm;
    float w2e = Wex2[c], w2a = Wam2[c];
    float b1e = bex1[c], b1a = bam1[c];
#pragma unroll
    for (int v = 0; v < 8; ++v) {
      float he = accE[t][v] + b1e; he = he > 0.f ? he : 0.f;
      float ha = accA[t][v] + b1a; ha = ha > 0.f ? ha : 0.f;
      rowE[v] += he * w2e;
      rowA[v] += ha * w2a;
    }
  }
#pragma unroll
  for (int off = 8; off; off >>= 1) {
#pragma unroll
    for (int v = 0; v < 8; ++v) {
      rowE[v] += __shfl_xor(rowE[v], off, 16);   // reduce within each 16-lane half
      rowA[v] += __shfl_xor(rowA[v], off, 16);
    }
  }
  if (lm == 0) {
    float b2e = bex2[0], b2a = bam2[0];
    int roff = hi ? 8 : 0;                       // D rows: v + 8*half -> edge e0+row
#pragma unroll
    for (int v = 0; v < 8; ++v) {
      size_t e = (size_t)(e0 + roff + v);
      out[2 * e]     = rowE[v] + b2e;
      out[2 * e + 1] = rowA[v] + b2a;
    }
  }
}

// =====================================================================
extern "C" void kernel_launch(void* const* d_in, const int* in_sizes, int n_in,
                              void* d_out, int out_size, void* d_ws, size_t ws_size,
                              hipStream_t stream) {
  const int* node_ids  = (const int*)d_in[0];
  const int* node_type = (const int*)d_in[1];
  const int* ei        = (const int*)d_in[2];
  const float* edge_attr = (const float*)d_in[3];
  const float* node_emb  = (const float*)d_in[4];
  const float* type_emb  = (const float*)d_in[5];
  const float* Wl1 = (const float*)d_in[6];  const float* bl1 = (const float*)d_in[7];
  const float* Wr1 = (const float*)d_in[8];  const float* br1 = (const float*)d_in[9];
  const float* We1 = (const float*)d_in[10]; const float* att1 = (const float*)d_in[11];
  const float* bias1 = (const float*)d_in[12];
  const float* Wl2 = (const float*)d_in[13]; const float* bl2 = (const float*)d_in[14];
  const float* Wr2 = (const float*)d_in[15]; const float* br2 = (const float*)d_in[16];
  const float* We2 = (const float*)d_in[17]; const float* att2 = (const float*)d_in[18];
  const float* bias2 = (const float*)d_in[19];
  const float* Wres1 = (const float*)d_in[20]; const float* Wres2 = (const float*)d_in[21];
  const float* g1 = (const float*)d_in[22]; const float* be1 = (const float*)d_in[23];
  const float* g2 = (const float*)d_in[24]; const float* be2 = (const float*)d_in[25];
  const float* Wex1 = (const float*)d_in[26]; const float* bex1 = (const float*)d_in[27];
  const float* Wex2 = (const float*)d_in[28]; const float* bex2 = (const float*)d_in[29];
  const float* Wam1 = (const float*)d_in[30]; const float* bam1 = (const float*)d_in[31];
  const float* Wam2 = (const float*)d_in[32]; const float* bam2 = (const float*)d_in[33];
  float* out = (float*)d_out;

  const int N = in_sizes[0];
  const int E = in_sizes[2] / 2;
  const int Etot = E + N;

  // ---- workspace bump allocator (16B-aligned chunks) ----
  size_t off = 0;
  auto alloc = [&](size_t bytes) {
    void* p = (char*)d_ws + off;
    off += (bytes + 15) & ~(size_t)15;
    return p;
  };
  __bf16* h0b   = (__bf16*)alloc((size_t)N * 64 * 2);
  __bf16* h1b   = (__bf16*)alloc((size_t)N * 256 * 2);
  __bf16* h2b   = (__bf16*)alloc((size_t)N * 64 * 2);
  float* bufXL = (float*)alloc((size_t)N * 256 * 4);   // xl (layer1 / reused layer2)
  float* bufXR = (float*)alloc((size_t)N * 256 * 4);   // xr
  float* bufRS = (float*)alloc((size_t)N * 256 * 4);   // residual GEMM
  float* bufMA = (float*)alloc((size_t)N * 256 * 4);   // message accumulator
  float* scbuf = (float*)alloc((size_t)Etot * 4 * 4);  // logits->exp (H=4 / H=1 reuse)
  unsigned* maxb = (unsigned*)alloc((size_t)N * 4 * 4);
  float* denb  = (float*)alloc((size_t)N * 4 * 4);
  // packed bf16 weights
  __bf16* Wl1p   = (__bf16*)alloc(64 * 256 * 2);
  __bf16* Wr1p   = (__bf16*)alloc(64 * 256 * 2);
  __bf16* Wres1p = (__bf16*)alloc(64 * 256 * 2);
  __bf16* Wl2p   = (__bf16*)alloc(256 * 64 * 2);
  __bf16* Wr2p   = (__bf16*)alloc(256 * 64 * 2);
  __bf16* Wres2p = (__bf16*)alloc(256 * 64 * 2);
  __bf16* Wex1p  = (__bf16*)alloc(128 * 64 * 2);
  __bf16* Wam1p  = (__bf16*)alloc(128 * 64 * 2);
  float* eam   = (float*)alloc(4);
  (void)ws_size; (void)n_in; (void)out_size;

  const int BT = 256;                      // 8 waves/block
  auto wave_blocks = [](long long waves) { return (unsigned)((waves + 7) / 8); };

  // prep: embeddings, edge-attr mean, weight packing
  k_h0<<<(N * 64 + BT - 1) / BT, BT, 0, stream>>>(node_ids, node_type, node_emb, type_emb,
                                                  h0b, N);
  k_eamean<<<1, 256, 0, stream>>>(edge_attr, eam, E);
  k_pack_w<<<64, BT, 0, stream>>>(Wl1,   Wl1p,   64, 256);
  k_pack_w<<<64, BT, 0, stream>>>(Wr1,   Wr1p,   64, 256);
  k_pack_w<<<64, BT, 0, stream>>>(Wres1, Wres1p, 64, 256);
  k_pack_w<<<64, BT, 0, stream>>>(Wl2,   Wl2p,   256, 64);
  k_pack_w<<<64, BT, 0, stream>>>(Wr2,   Wr2p,   256, 64);
  k_pack_w<<<64, BT, 0, stream>>>(Wres2, Wres2p, 256, 64);
  k_pack_w<<<32, BT, 0, stream>>>(Wex1,  Wex1p,  128, 64);
  k_pack_w<<<32, BT, 0, stream>>>(Wam1,  Wam1p,  128, 64);

  // ---------------- layer 1 (H=4, HC=256) ----------------
  {
    long long waves = (long long)(N / 16) * (256 / 64);
    k_gemm_wmma<<<wave_blocks(waves), BT, 0, stream>>>(h0b, Wl1p,  bl1,     bufXL, N, 64, 256);
    k_gemm_wmma<<<wave_blocks(waves), BT, 0, stream>>>(h0b, Wr1p,  br1,     bufXR, N, 64, 256);
    k_gemm_wmma<<<wave_blocks(waves), BT, 0, stream>>>(h0b, Wres1p, nullptr, bufRS, N, 64, 256);
  }
  k_fill_u32<<<2048, BT, 0, stream>>>(maxb, ENC_NEG_INF, (long long)N * 4);
  k_fill_u32<<<2048, BT, 0, stream>>>((unsigned*)denb, 0u, (long long)N * 4);
  k_fill_u32<<<2048, BT, 0, stream>>>((unsigned*)bufMA, 0u, (long long)N * 256);
  k_att_logits<<<wave_blocks(Etot), BT, 0, stream>>>(bufXL, bufXR, We1, att1, ei, edge_attr,
                                                     eam, maxb, scbuf, E, N, 4);
  k_att_exp<<<(unsigned)(((long long)Etot * 4 + BT - 1) / BT), BT, 0, stream>>>(
      ei, maxb, scbuf, denb, E, N, 2);
  k_att_scatter<<<wave_blocks(Etot), BT, 0, stream>>>(bufXL, ei, scbuf, denb, bufMA, E, N, 4);
  k_elu_res_ln<<<wave_blocks(N), BT, 0, stream>>>(bufRS, bufMA, bias1, g1, be1, h1b, N, 256);

  // ---------------- layer 2 (H=1, HC=64) ----------------
  {
    long long waves = (long long)(N / 16) * (64 / 64);
    k_gemm_wmma<<<wave_blocks(waves), BT, 0, stream>>>(h1b, Wl2p,  bl2,     bufXL, N, 256, 64);
    k_gemm_wmma<<<wave_blocks(waves), BT, 0, stream>>>(h1b, Wr2p,  br2,     bufXR, N, 256, 64);
    k_gemm_wmma<<<wave_blocks(waves), BT, 0, stream>>>(h1b, Wres2p, nullptr, bufRS, N, 256, 64);
  }
  k_fill_u32<<<2048, BT, 0, stream>>>(maxb, ENC_NEG_INF, (long long)N);
  k_fill_u32<<<2048, BT, 0, stream>>>((unsigned*)denb, 0u, (long long)N);
  k_fill_u32<<<2048, BT, 0, stream>>>((unsigned*)bufMA, 0u, (long long)N * 64);
  k_att_logits<<<wave_blocks(Etot), BT, 0, stream>>>(bufXL, bufXR, We2, att2, ei, edge_attr,
                                                     eam, maxb, scbuf, E, N, 1);
  k_att_exp<<<(unsigned)(((long long)Etot + BT - 1) / BT), BT, 0, stream>>>(
      ei, maxb, scbuf, denb, E, N, 0);
  k_att_scatter<<<wave_blocks(Etot), BT, 0, stream>>>(bufXL, ei, scbuf, denb, bufMA, E, N, 1);
  k_elu_res_ln<<<wave_blocks(N), BT, 0, stream>>>(bufRS, bufMA, bias2, g2, be2, h2b, N, 64);

  // ---------------- edge-pair MLP head (WMMA, fused both heads) ----------------
  k_edge_mlp<<<wave_blocks(E / 16), BT, 0, stream>>>(h2b, ei, Wex1p, bex1, Wex2, bex2,
                                                     Wam1p, bam1, Wam2, bam2, out, E);
}
